// EquiformerBlock_72662256713902
// MI455X (gfx1250) — compile-verified, matrix operand order checked by reading, not run on hardware
//
#include <hip/hip_runtime.h>
#include <hip/hip_bf16.h>

// Problem constants (match reference)
#define NSN 50000
#define NDN 50000
#define EN  600000
// C=64, A=16, S=64, H=8, DH=8, M=192

typedef __attribute__((ext_vector_type(16))) _Float16 v16h;
typedef __attribute__((ext_vector_type(8)))  float    v8f;

__device__ __forceinline__ v8f wmma16(v16h a, v16h b, v8f c) {
  return __builtin_amdgcn_wmma_f32_16x16x32_f16(false, a, false, b, (short)0, c, false, false);
}

// A-operand (16x32 f16 tile) loaded from row-major f32 global memory.
// lane&15 = row, lane>>4 selects K-halves {8h..8h+7, 16+8h..23+8h}.
__device__ __forceinline__ v16h load_a_g32(const float* __restrict__ src, int stride, int k0, int lane) {
  int r = lane & 15, h = lane >> 4;
  const float* p = src + (size_t)r * stride + k0 + 8 * h;
  v16h a;
#pragma unroll
  for (int i = 0; i < 8; ++i) { a[i] = (_Float16)p[i]; a[8 + i] = (_Float16)p[16 + i]; }
  return a;
}

// A-operand from row-major f16 LDS stage.
__device__ __forceinline__ v16h load_a_lds(const _Float16* s, int stride, int k0, int lane) {
  int r = lane & 15, h = lane >> 4;
  const _Float16* p = s + r * stride + k0 + 8 * h;
  v16h a;
#pragma unroll
  for (int i = 0; i < 8; ++i) { a[i] = p[i]; a[8 + i] = p[16 + i]; }
  return a;
}

__device__ __forceinline__ float siluf(float x) { return x * (1.f / (1.f + __expf(-x))); }

__device__ __forceinline__ void atomicMaxF(float* a, float v) {
  if (v >= 0.f) atomicMax((int*)a, __float_as_int(v));
  else          atomicMin((unsigned int*)a, __float_as_uint(v));
}

// ---------------------------------------------------------------------------
// Pack a row-major f32 weight [Kvalid x N] into per-lane B-operand f16 layout.
// dst[((chunk*tiles + tile)*32 + lane)*16 + i] = W[k][tile*16 + (lane&15)],
// k = chunk*32 + (i<8 ? i : i+8) + 8*(lane>>4); rows >= Kvalid are zero-padded.
__global__ void pack_b_kernel(const float* __restrict__ W, int Kvalid, int N, int total,
                              _Float16* __restrict__ dst) {
  int idx = blockIdx.x * blockDim.x + threadIdx.x;
  if (idx >= total) return;
  int i = idx & 15;
  int lane = (idx >> 4) & 31;
  int ct = idx >> 9;
  int tiles = N >> 4;
  int tile = ct % tiles;
  int chunk = ct / tiles;
  int h = lane >> 4;
  int k = chunk * 32 + (i < 8 ? i : i + 8) + 8 * h;
  int n = tile * 16 + (lane & 15);
  float v = (k < Kvalid) ? W[(size_t)k * N + n] : 0.f;
  dst[idx] = (_Float16)v;
}

// ---------------------------------------------------------------------------
__global__ void init_ws_kernel(float* __restrict__ node_max, float* __restrict__ node_sum,
                               float* __restrict__ out_acc) {
  int idx = blockIdx.x * blockDim.x + threadIdx.x;
  if (idx < NDN * 8) { node_max[idx] = -__builtin_inff(); node_sum[idx] = 0.f; }
  if (idx < NDN * 64) out_acc[idx] = 0.f;
}

// ---------------------------------------------------------------------------
// Node projection: Y = LN(X) @ W (+bias). 1 wave = 16 rows. 625 blocks x 5 waves.
__global__ __launch_bounds__(160) void node_proj_kernel(
    const float* __restrict__ X, const float* __restrict__ lnw, const float* __restrict__ lnb,
    const _Float16* __restrict__ Bp, const float* __restrict__ bias, float* __restrict__ Y) {
  __shared__ _Float16 sh[5][16 * 64];
  int wave = threadIdx.x >> 5, lane = threadIdx.x & 31;
  int tile = blockIdx.x * 5 + wave;
  int r0 = tile * 16;
  int n = lane & 15, h = lane >> 4;

  // LayerNorm: 2 lanes per row, 32 cols each, combine via shfl_xor(16).
  {
    const float* p = X + (size_t)(r0 + n) * 64 + 32 * h;
    float s = 0.f, sq = 0.f, xv[32];
#pragma unroll
    for (int i = 0; i < 32; ++i) { float x = p[i]; xv[i] = x; s += x; sq += x * x; }
    s += __shfl_xor(s, 16);
    sq += __shfl_xor(sq, 16);
    float mean = s * (1.f / 64.f);
    float var = sq * (1.f / 64.f) - mean * mean;
    float rs = __frsqrt_rn(var + 1e-5f);
#pragma unroll
    for (int i = 0; i < 32; ++i) {
      int col = 32 * h + i;
      float y = (xv[i] - mean) * rs * lnw[col] + lnb[col];
      sh[wave][n * 64 + col] = (_Float16)y;
    }
  }
  __syncthreads();

  v8f acc[4] = {};
#pragma unroll
  for (int chunk = 0; chunk < 2; ++chunk) {
    v16h a = load_a_lds(&sh[wave][0], 64, chunk * 32, lane);
#pragma unroll
    for (int t = 0; t < 4; ++t) {
      v16h b = *(const v16h*)(Bp + ((size_t)(chunk * 4 + t) * 32 + lane) * 16);
      acc[t] = wmma16(a, b, acc[t]);
    }
  }
#pragma unroll
  for (int t = 0; t < 4; ++t) {
    int col = t * 16 + n;
    float bb = bias ? bias[col] : 0.f;
#pragma unroll
    for (int v = 0; v < 8; ++v) {
      int row = r0 + v + 8 * h;
      Y[(size_t)row * 64 + col] = acc[t][v] + bb;
    }
  }
}

// ---------------------------------------------------------------------------
// Fused per-edge pipeline. 1 wave = 16 edges. 9375 blocks x 4 waves = 37500 tiles.
__global__ __launch_bounds__(128) void edge_kernel(
    const float* __restrict__ edge_attr, const float* __restrict__ edge_scalars,
    const float* __restrict__ edge_logits, const int* __restrict__ edge_src,
    const int* __restrict__ edge_dst, const float* __restrict__ ms, const float* __restrict__ md,
    const float* __restrict__ b_r1, const float* __restrict__ b_r2,
    const float* __restrict__ alpha_vec,
    const _Float16* __restrict__ Bp_r1, const _Float16* __restrict__ Bp_r2,
    const _Float16* __restrict__ Bp_ea, const _Float16* __restrict__ Bp_dtp,
    float* __restrict__ logits_ws, _Float16* __restrict__ val_ws, float* __restrict__ node_max) {
  __shared__ _Float16 sh_h[4][16 * 64];
  __shared__ float sh_f[4][16 * 64];
  int wave = threadIdx.x >> 5, lane = threadIdx.x & 31;
  size_t e0 = ((size_t)blockIdx.x * 4 + wave) * 16;
  int n = lane & 15, h = lane >> 4;

  // 1) r = silu(edge_scalars @ W_r1 + b_r1)
  v8f racc[4] = {};
#pragma unroll
  for (int chunk = 0; chunk < 2; ++chunk) {
    v16h a = load_a_g32(edge_scalars + e0 * 64, 64, chunk * 32, lane);
#pragma unroll
    for (int t = 0; t < 4; ++t) {
      v16h b = *(const v16h*)(Bp_r1 + ((size_t)(chunk * 4 + t) * 32 + lane) * 16);
      racc[t] = wmma16(a, b, racc[t]);
    }
  }
#pragma unroll
  for (int t = 0; t < 4; ++t) {
    int col = t * 16 + n;
    float bb = b_r1[col];
#pragma unroll
    for (int v = 0; v < 8; ++v)
      sh_h[wave][(v + 8 * h) * 64 + col] = (_Float16)siluf(racc[t][v] + bb);
  }
  __syncthreads();

  // 2) w_edge = r @ W_r2 + b_r2  (kept in registers, D layout)
  v8f wacc[8] = {};
#pragma unroll
  for (int chunk = 0; chunk < 2; ++chunk) {
    v16h a = load_a_lds(&sh_h[wave][0], 64, chunk * 32, lane);
#pragma unroll
    for (int t = 0; t < 8; ++t) {
      v16h b = *(const v16h*)(Bp_r2 + ((size_t)(chunk * 8 + t) * 32 + lane) * 16);
      wacc[t] = wmma16(a, b, wacc[t]);
    }
  }
#pragma unroll
  for (int t = 0; t < 8; ++t) {
    float bb = b_r2[t * 16 + n];
#pragma unroll
    for (int v = 0; v < 8; ++v) wacc[t][v] += bb;
  }
  __syncthreads();  // done reading sh_h (rewritten in step 4)

  // 3) ea = edge_attr @ W_ea  (K=16 zero-padded to 32) -> sh_f (f32 row-major)
  {
    v16h a;
    const float* p = edge_attr + (e0 + n) * 16 + 8 * h;
#pragma unroll
    for (int i = 0; i < 8; ++i) { a[i] = (_Float16)p[i]; a[8 + i] = (_Float16)0.f; }
    v8f eacc[4] = {};
#pragma unroll
    for (int t = 0; t < 4; ++t) {
      v16h b = *(const v16h*)(Bp_ea + ((size_t)t * 32 + lane) * 16);
      eacc[t] = wmma16(a, b, eacc[t]);
    }
#pragma unroll
    for (int t = 0; t < 4; ++t)
#pragma unroll
      for (int v = 0; v < 8; ++v)
        sh_f[wave][(v + 8 * h) * 64 + t * 16 + n] = eacc[t][v];
  }
  __syncthreads();

  // 4) prod = (ms[src] + md[dst]) * ea  -> sh_h (f16). Gathers hit L2 (node tables are 12.8MB).
  {
    size_t e = e0 + n;
    int si = edge_src[e], di = edge_dst[e];
    const float* pms = ms + (size_t)si * 64 + 32 * h;
    const float* pmd = md + (size_t)di * 64 + 32 * h;
    const float* pea = &sh_f[wave][n * 64 + 32 * h];
    _Float16* po = &sh_h[wave][n * 64 + 32 * h];
#pragma unroll
    for (int i = 0; i < 32; ++i) po[i] = (_Float16)((pms[i] + pmd[i]) * pea[i]);
  }
  __syncthreads();

  // 5) dtp = (prod @ W_dtp) * w_edge
  v8f dacc[8] = {};
#pragma unroll
  for (int chunk = 0; chunk < 2; ++chunk) {
    v16h a = load_a_lds(&sh_h[wave][0], 64, chunk * 32, lane);
#pragma unroll
    for (int t = 0; t < 8; ++t) {
      v16h b = *(const v16h*)(Bp_dtp + ((size_t)(chunk * 8 + t) * 32 + lane) * 16);
      dacc[t] = wmma16(a, b, dacc[t]);
    }
  }
#pragma unroll
  for (int t = 0; t < 8; ++t) dacc[t] = dacc[t] * wacc[t];

  // 6) logits: sum_d leaky_relu(alpha_feat)*alpha_vec, reduce across 8 lanes (d = lane&7).
#pragma unroll
  for (int t = 0; t < 4; ++t) {
    int col = t * 16 + n;
    int head = col >> 3;
    float av = alpha_vec[head * 8 + (col & 7)];
#pragma unroll
    for (int v = 0; v < 8; ++v) {
      float x = dacc[t][v];
      float p = (x > 0.f ? x : 0.2f * x) * av;
      p += __shfl_xor(p, 1);
      p += __shfl_xor(p, 2);
      p += __shfl_xor(p, 4);
      if ((n & 7) == 0) {
        size_t e = e0 + v + 8 * h;
        float logit = p + edge_logits[e];
        logits_ws[e * 8 + head] = logit;
        atomicMaxF(&node_max[(size_t)edge_dst[e] * 8 + head], logit);
      }
    }
  }
  // val half of dtp -> f16 workspace
#pragma unroll
  for (int t = 4; t < 8; ++t) {
    int c = (t - 4) * 16 + n;
#pragma unroll
    for (int v = 0; v < 8; ++v)
      val_ws[(e0 + v + 8 * h) * 64 + c] = (_Float16)dacc[t][v];
  }
}

// ---------------------------------------------------------------------------
__global__ void softmax_pass2_kernel(const int* __restrict__ edge_dst,
                                     const float* __restrict__ node_max,
                                     float* __restrict__ logits_ws,
                                     float* __restrict__ node_sum) {
  size_t idx = (size_t)blockIdx.x * blockDim.x + threadIdx.x;
  if (idx >= (size_t)EN * 8) return;
  size_t e = idx >> 3;
  int hh = idx & 7;
  int d = edge_dst[e];
  float a = __expf(logits_ws[idx] - node_max[(size_t)d * 8 + hh]);
  logits_ws[idx] = a;
  atomicAdd(&node_sum[(size_t)d * 8 + hh], a);
}

__global__ void scatter_pass3_kernel(const int* __restrict__ edge_dst,
                                     const float* __restrict__ logits_ws,
                                     const float* __restrict__ node_sum,
                                     const _Float16* __restrict__ val_ws,
                                     float* __restrict__ out_acc) {
  size_t idx = (size_t)blockIdx.x * blockDim.x + threadIdx.x;
  if (idx >= (size_t)EN * 64) return;
  size_t e = idx >> 6;
  int c = idx & 63;
  int hh = c >> 3;
  int d = edge_dst[e];
  float attn = logits_ws[e * 8 + hh] / node_sum[(size_t)d * 8 + hh];
  atomicAdd(&out_acc[(size_t)d * 64 + c], (float)val_ws[idx] * attn);
}

// ---------------------------------------------------------------------------
// Node finalize: proj + residual + postLN + FFN + residual. 625 blocks x 5 waves.
__global__ __launch_bounds__(160) void node_final_kernel(
    const float* __restrict__ out_acc, const float* __restrict__ dst_f,
    const float* __restrict__ b_proj, const float* __restrict__ ln_w,
    const float* __restrict__ ln_b, const float* __restrict__ b_f1,
    const float* __restrict__ b_f2, const _Float16* __restrict__ Bp_proj,
    const _Float16* __restrict__ Bp_f1, const _Float16* __restrict__ Bp_f2,
    float* __restrict__ out) {
  __shared__ float    sh_f[5][16 * 64];
  __shared__ _Float16 sh_y[5][16 * 64];
  __shared__ _Float16 sh_g[5][16 * 192];
  int wave = threadIdx.x >> 5, lane = threadIdx.x & 31;
  int r0 = (blockIdx.x * 5 + wave) * 16;
  int n = lane & 15, h = lane >> 4;

  // out = out_acc @ W_proj + b_proj ; emb = out + dst_f
  v8f acc[4] = {};
#pragma unroll
  for (int chunk = 0; chunk < 2; ++chunk) {
    v16h a = load_a_g32(out_acc + (size_t)r0 * 64, 64, chunk * 32, lane);
#pragma unroll
    for (int t = 0; t < 4; ++t) {
      v16h b = *(const v16h*)(Bp_proj + ((size_t)(chunk * 4 + t) * 32 + lane) * 16);
      acc[t] = wmma16(a, b, acc[t]);
    }
  }
  v8f emb[4];
#pragma unroll
  for (int t = 0; t < 4; ++t) {
    int col = t * 16 + n;
    float bb = b_proj[col];
#pragma unroll
    for (int v = 0; v < 8; ++v) {
      int row = v + 8 * h;
      float e_ = acc[t][v] + bb + dst_f[(size_t)(r0 + row) * 64 + col];
      emb[t][v] = e_;
      sh_f[wave][row * 64 + col] = e_;
    }
  }
  __syncthreads();

  // post LayerNorm -> sh_y (f16)
  {
    const float* p = &sh_f[wave][n * 64 + 32 * h];
    float s = 0.f, sq = 0.f, xv[32];
#pragma unroll
    for (int i = 0; i < 32; ++i) { float x = p[i]; xv[i] = x; s += x; sq += x * x; }
    s += __shfl_xor(s, 16);
    sq += __shfl_xor(sq, 16);
    float mean = s * (1.f / 64.f);
    float var = sq * (1.f / 64.f) - mean * mean;
    float rs = __frsqrt_rn(var + 1e-5f);
#pragma unroll
    for (int i = 0; i < 32; ++i) {
      int col = 32 * h + i;
      sh_y[wave][n * 64 + col] = (_Float16)((xv[i] - mean) * rs * ln_w[col] + ln_b[col]);
    }
  }
  __syncthreads();

  // FFN up: [16,64] @ [64,192], silu -> sh_g
  v8f gacc[12] = {};
#pragma unroll
  for (int chunk = 0; chunk < 2; ++chunk) {
    v16h a = load_a_lds(&sh_y[wave][0], 64, chunk * 32, lane);
#pragma unroll
    for (int t = 0; t < 12; ++t) {
      v16h b = *(const v16h*)(Bp_f1 + ((size_t)(chunk * 12 + t) * 32 + lane) * 16);
      gacc[t] = wmma16(a, b, gacc[t]);
    }
  }
#pragma unroll
  for (int t = 0; t < 12; ++t) {
    int col = t * 16 + n;
    float bb = b_f1[col];
#pragma unroll
    for (int v = 0; v < 8; ++v)
      sh_g[wave][(v + 8 * h) * 192 + col] = (_Float16)siluf(gacc[t][v] + bb);
  }
  __syncthreads();

  // FFN down: [16,192] @ [192,64] + b_f2 + emb
  v8f facc[4] = {};
#pragma unroll
  for (int chunk = 0; chunk < 6; ++chunk) {
    v16h a = load_a_lds(&sh_g[wave][0], 192, chunk * 32, lane);
#pragma unroll
    for (int t = 0; t < 4; ++t) {
      v16h b = *(const v16h*)(Bp_f2 + ((size_t)(chunk * 4 + t) * 32 + lane) * 16);
      facc[t] = wmma16(a, b, facc[t]);
    }
  }
#pragma unroll
  for (int t = 0; t < 4; ++t) {
    int col = t * 16 + n;
    float bb = b_f2[col];
#pragma unroll
    for (int v = 0; v < 8; ++v) {
      int row = v + 8 * h;
      out[(size_t)(r0 + row) * 64 + col] = facc[t][v] + bb + emb[t][v];
    }
  }
}

// ---------------------------------------------------------------------------
extern "C" void kernel_launch(void* const* d_in, const int* in_sizes, int n_in,
                              void* d_out, int out_size, void* d_ws, size_t ws_size,
                              hipStream_t stream) {
  const float* src_f        = (const float*)d_in[0];
  const float* dst_f        = (const float*)d_in[1];
  const float* edge_attr    = (const float*)d_in[2];
  const float* edge_scalars = (const float*)d_in[3];
  const float* edge_logits  = (const float*)d_in[4];
  const int*   edge_src     = (const int*)d_in[5];
  const int*   edge_dst     = (const int*)d_in[6];
  const float* ln_src_w = (const float*)d_in[7];
  const float* ln_src_b = (const float*)d_in[8];
  const float* W_src    = (const float*)d_in[9];
  const float* ln_dst_w = (const float*)d_in[10];
  const float* ln_dst_b = (const float*)d_in[11];
  const float* W_dst    = (const float*)d_in[12];
  const float* b_dst    = (const float*)d_in[13];
  const float* W_ea     = (const float*)d_in[14];
  const float* W_dtp    = (const float*)d_in[15];
  const float* W_r1     = (const float*)d_in[16];
  const float* b_r1     = (const float*)d_in[17];
  const float* W_r2     = (const float*)d_in[18];
  const float* b_r2     = (const float*)d_in[19];
  const float* alpha_vec = (const float*)d_in[20];
  const float* W_proj   = (const float*)d_in[21];
  const float* b_proj   = (const float*)d_in[22];
  const float* ln_post_w = (const float*)d_in[23];
  const float* ln_post_b = (const float*)d_in[24];
  const float* W_f1     = (const float*)d_in[25];
  const float* b_f1     = (const float*)d_in[26];
  const float* W_f2     = (const float*)d_in[27];
  const float* b_f2     = (const float*)d_in[28];
  float* out = (float*)d_out;

  // Workspace layout
  float* ws_ms   = (float*)d_ws;                       // [NS,64] f32
  float* ws_md   = ws_ms + (size_t)NSN * 64;           // [ND,64] f32
  float* ws_out  = ws_md + (size_t)NDN * 64;           // [ND,64] f32 (attention output accum)
  float* ws_nmax = ws_out + (size_t)NDN * 64;          // [ND,8]
  float* ws_nsum = ws_nmax + (size_t)NDN * 8;          // [ND,8]
  float* ws_log  = ws_nsum + (size_t)NDN * 8;          // [E,8]
  _Float16* ws_val = (_Float16*)(ws_log + (size_t)EN * 8);  // [E,64] f16
  _Float16* pw = ws_val + (size_t)EN * 64;             // packed f16 weights
  _Float16* pw_src  = pw;
  _Float16* pw_dst  = pw_src + 4096;
  _Float16* pw_ea   = pw_dst + 4096;   // padded 32x64
  _Float16* pw_r1   = pw_ea + 2048;
  _Float16* pw_r2   = pw_r1 + 4096;
  _Float16* pw_dtp  = pw_r2 + 8192;
  _Float16* pw_proj = pw_dtp + 8192;
  _Float16* pw_f1   = pw_proj + 4096;
  _Float16* pw_f2   = pw_f1 + 12288;

  // Pack weights into WMMA B-operand layout (f16)
  auto pack = [&](const float* W, int Kvalid, int N, int Kpad, _Float16* dst) {
    int total = Kpad * N;
    pack_b_kernel<<<(total + 255) / 256, 256, 0, stream>>>(W, Kvalid, N, total, dst);
  };
  pack(W_src, 64, 64, 64, pw_src);
  pack(W_dst, 64, 64, 64, pw_dst);
  pack(W_ea, 16, 64, 32, pw_ea);
  pack(W_r1, 64, 64, 64, pw_r1);
  pack(W_r2, 64, 128, 64, pw_r2);
  pack(W_dtp, 64, 128, 64, pw_dtp);
  pack(W_proj, 64, 64, 64, pw_proj);
  pack(W_f1, 64, 192, 64, pw_f1);
  pack(W_f2, 192, 64, 192, pw_f2);

  // Init accumulators
  init_ws_kernel<<<(NDN * 64 + 255) / 256, 256, 0, stream>>>(ws_nmax, ws_nsum, ws_out);

  // Node projections (3125 tiles = 625 blocks x 5 waves)
  node_proj_kernel<<<625, 160, 0, stream>>>(src_f, ln_src_w, ln_src_b, pw_src, nullptr, ws_ms);
  node_proj_kernel<<<625, 160, 0, stream>>>(dst_f, ln_dst_w, ln_dst_b, pw_dst, b_dst, ws_md);

  // Fused edge pipeline (37500 tiles = 9375 blocks x 4 waves)
  edge_kernel<<<9375, 128, 0, stream>>>(edge_attr, edge_scalars, edge_logits, edge_src, edge_dst,
                                        ws_ms, ws_md, b_r1, b_r2, alpha_vec,
                                        pw_r1, pw_r2, pw_ea, pw_dtp,
                                        ws_log, ws_val, ws_nmax);

  // Segment softmax phases
  softmax_pass2_kernel<<<(EN * 8) / 256, 256, 0, stream>>>(edge_dst, ws_nmax, ws_log, ws_nsum);
  scatter_pass3_kernel<<<((size_t)EN * 64) / 256, 256, 0, stream>>>(edge_dst, ws_log, ws_nsum,
                                                                    ws_val, ws_out);

  // Node finalize (proj + residual + LN + FFN + residual)
  node_final_kernel<<<625, 160, 0, stream>>>(ws_out, dst_f, b_proj, ln_post_w, ln_post_b,
                                             b_f1, b_f2, pw_proj, pw_f1, pw_f2, out);
  (void)in_sizes; (void)n_in; (void)out_size; (void)ws_size;
}
